// SelectiveStateSpaceModel_20177756357189
// MI455X (gfx1250) — compile-verified, matrix operand order checked by reading, not run on hardware
//
#include <hip/hip_runtime.h>
#include <hip/hip_bf16.h>
#include <math.h>
#include <stdint.h>

#define BATCH   2
#define SEQ     2048
#define DMODEL  1024
#define DSTATE  16
#define DCONV   4
#define DINNER  2048
#define DTRANK  64
#define NROWS   (BATCH*SEQ)          /* 4096 */
#define XDBLW   (DTRANK + 2*DSTATE)  /* 96 */

typedef __bf16 bf16;
typedef __attribute__((ext_vector_type(16))) bf16  v16bf;
typedef __attribute__((ext_vector_type(8)))  bf16  v8bf;
typedef __attribute__((ext_vector_type(8)))  float v8f;

// ---------------------------------------------------- CDNA5 async helpers --
// Async copy global -> LDS (tracked by ASYNCcnt). VDST operand is a VGPR
// holding the LDS byte offset (low 32 bits of the flat shared-space address).
__device__ __forceinline__ void async_copy_b128(void* lds_ptr, const void* gptr) {
  uint32_t loff = (uint32_t)(uintptr_t)lds_ptr;
  asm volatile("global_load_async_to_lds_b128 %0, %1, off"
               :: "v"(loff), "v"(gptr) : "memory");
}
__device__ __forceinline__ void wait_async_barrier() {
  asm volatile("s_wait_asynccnt 0x0" ::: "memory");
  __syncthreads();
}

__device__ __forceinline__ v16bf frag_from_lds(const bf16* p, int koff) {
  v8bf lo = *(const v8bf*)(p + koff);
  v8bf hi = *(const v8bf*)(p + 16 + koff);
  v16bf f;
#pragma unroll
  for (int i = 0; i < 8; ++i) { f[i] = lo[i]; f[i + 8] = hi[i]; }
  return f;
}

// ---------------------------------------------------------------- casts ----
__global__ void cast_f32_to_bf16(const float* __restrict__ src,
                                 bf16* __restrict__ dst, int n) {
  int i = blockIdx.x * blockDim.x + threadIdx.x;
  if (i < n) dst[i] = (bf16)src[i];
}

__global__ void cast_deltar(const float* __restrict__ xdbl,
                            bf16* __restrict__ dr) {
  int i = blockIdx.x * blockDim.x + threadIdx.x;
  if (i >= NROWS * DTRANK) return;
  int r = i / DTRANK, c = i % DTRANK;
  dr[i] = (bf16)xdbl[(size_t)r * XDBLW + c];
}

// --------------------------------- WMMA GEMM, LDS-staged via async copy ----
// C(M,N) = A(M,K) @ W(N,K)^T.  Block = 128 threads = 4 waves.
// Tile BM=128 x BN=64 x BK=64, double-buffered in LDS with
// global_load_async_to_lds_b128; wave computes 2 M-tiles x 4 N-tiles.
// Requires N%64==0, M%128==0, K%64==0.  MODE==1: softplus(acc+bias[n]).
template <int MODE>
__global__ __launch_bounds__(128)
void gemm_bf16_wmma_lds(const bf16* __restrict__ A, const bf16* __restrict__ W,
                        float* __restrict__ C, const float* __restrict__ bias,
                        int M, int N, int K) {
  __shared__ __align__(16) bf16 As[2][128 * 64];
  __shared__ __align__(16) bf16 Bs[2][64 * 64];

  const int tid    = threadIdx.x;
  const int lane   = tid & 31;
  const int wave   = tid >> 5;
  const int blockM = blockIdx.y * 128;
  const int blockN = blockIdx.x * 64;
  const int r      = lane & 15;
  const int koff   = (lane >> 4) * 8;   // ISA A/B fragment half-lane split

  auto stage_in = [&](int buf, int kb) {
#pragma unroll
    for (int j = 0; j < 8; ++j) {        // A tile: 128 rows x 64 cols
      int q = tid + 128 * j;
      int i = q >> 3, c = (q & 7) * 8;
      async_copy_b128(&As[buf][i * 64 + c],
                      A + (size_t)(blockM + i) * K + kb + c);
    }
#pragma unroll
    for (int j = 0; j < 4; ++j) {        // B tile: 64 rows x 64 cols
      int q = tid + 128 * j;
      int n = q >> 3, c = (q & 7) * 8;
      async_copy_b128(&Bs[buf][n * 64 + c],
                      W + (size_t)(blockN + n) * K + kb + c);
    }
  };

  v8f acc[2][4] = {};

  stage_in(0, 0);
  wait_async_barrier();

  int buf = 0;
  for (int kb = 0; kb < K; kb += 64) {
    if (kb + 64 < K) stage_in(buf ^ 1, kb + 64);   // prefetch next stage
#pragma unroll
    for (int ks = 0; ks < 2; ++ks) {               // two 32-K steps per stage
      v16bf af[2];
#pragma unroll
      for (int m = 0; m < 2; ++m)
        af[m] = frag_from_lds(&As[buf][(wave * 32 + m * 16 + r) * 64 + ks * 32],
                              koff);
#pragma unroll
      for (int t = 0; t < 4; ++t) {
        v16bf bfr =
            frag_from_lds(&Bs[buf][(t * 16 + r) * 64 + ks * 32], koff);
#pragma unroll
        for (int m = 0; m < 2; ++m)
          acc[m][t] = __builtin_amdgcn_wmma_f32_16x16x32_bf16(
              false, af[m], false, bfr, (short)0, acc[m][t], false, false);
      }
    }
    wait_async_barrier();   // next stage resident; all waves done reading buf
    buf ^= 1;
  }

  // C/D layout: VGPR j -> (M = j + (lane>=16)*8, N = lane&15)
#pragma unroll
  for (int m = 0; m < 2; ++m) {
    const int m0 = blockM + wave * 32 + m * 16 + (lane >> 4) * 8;
#pragma unroll
    for (int t = 0; t < 4; ++t) {
      const int n0 = blockN + t * 16 + (lane & 15);
#pragma unroll
      for (int j = 0; j < 8; ++j) {
        float v = acc[m][t][j];
        if (MODE == 1) {
          float s = v + bias[n0];
          v = (s > 20.f) ? s : log1pf(__expf(s));   // softplus
        }
        C[(size_t)(m0 + j) * N + n0] = v;
      }
    }
  }
}

// ------------------------- WMMA GEMM, direct from global (small N=96) ----
template <int NT>
__global__ __launch_bounds__(256)
void gemm_bf16_wmma_direct(const bf16* __restrict__ A,
                           const bf16* __restrict__ W,
                           float* __restrict__ C, int M, int N, int K) {
  const int lane = threadIdx.x & 31;
  const int wave = threadIdx.x >> 5;
  const int mt   = blockIdx.y * 8 + wave;
  const int ntb  = blockIdx.x * NT;
  const int r    = lane & 15;
  const int koff = (lane >> 4) * 8;

  const bf16* arow = A + (size_t)(mt * 16 + r) * K;
  v8f acc[NT] = {};

  for (int kb = 0; kb < K; kb += 32) {
    v16bf af = frag_from_lds(arow + kb, koff);   // same pattern from global
#pragma unroll
    for (int t = 0; t < NT; ++t) {
      const bf16* wrow = W + (size_t)((ntb + t) * 16 + r) * K;
      v16bf bfr = frag_from_lds(wrow + kb, koff);
      acc[t] = __builtin_amdgcn_wmma_f32_16x16x32_bf16(
          false, af, false, bfr, (short)0, acc[t], false, false);
    }
  }
  const int m0 = mt * 16 + (lane >> 4) * 8;
#pragma unroll
  for (int t = 0; t < NT; ++t) {
    const int n0 = (ntb + t) * 16 + (lane & 15);
#pragma unroll
    for (int j = 0; j < 8; ++j)
      C[(size_t)(m0 + j) * N + n0] = acc[t][j];
  }
}

// --------------------------------------- causal depthwise conv + SiLU ----
__global__ void conv_silu_kernel(const float* __restrict__ xz,
                                 const float* __restrict__ cw,
                                 const float* __restrict__ cb,
                                 float* __restrict__ u,
                                 bf16* __restrict__ ubf) {
  int idx = blockIdx.x * blockDim.x + threadIdx.x;
  if (idx >= NROWS * DINNER) return;
  int d   = idx % DINNER;
  int row = idx / DINNER;           // b*SEQ + t
  int t   = row % SEQ;
  float acc = cb[d];
#pragma unroll
  for (int j = 0; j < DCONV; ++j) {
    int ts = t - (DCONV - 1) + j;
    if (ts >= 0)
      acc += cw[d * DCONV + j] *
             xz[(size_t)(row - (DCONV - 1) + j) * (2 * DINNER) + d];
  }
  float s = acc / (1.f + __expf(-acc));
  u[idx]   = s;
  ubf[idx] = (bf16)s;
}

// ------------------------------------------------------ selective scan ----
// 16 lanes per (b,d) channel (one state element each); wave32 = 2 channels.
// y reduced with 4 shfl_xor steps; fuses +u*D skip and *silu(z) gate.
__global__ __launch_bounds__(256)
void scan_kernel(const float* __restrict__ delta, const float* __restrict__ u,
                 const float* __restrict__ xdbl, const float* __restrict__ xz,
                 const float* __restrict__ A_log, const float* __restrict__ Dp,
                 bf16* __restrict__ ybf) {
  const int n = threadIdx.x & 15;
  const int g = threadIdx.x >> 4;
  const int d = blockIdx.x * 16 + g;
  const int b = blockIdx.y;

  const float A  = -__expf(A_log[d * DSTATE + n]);
  const float Dd = Dp[d];
  float h = 0.f;

  const size_t rowbase = (size_t)b * SEQ;
  for (int t = 0; t < SEQ; ++t) {
    const size_t row = rowbase + t;
    float dv = delta[row * DINNER + d];
    float uv = u[row * DINNER + d];
    float Bn = xdbl[row * XDBLW + DTRANK + n];
    float Cn = xdbl[row * XDBLW + DTRANK + DSTATE + n];

    float dA = __expf(dv * A);
    h = dA * h + dv * Bn * uv;
    float yv = h * Cn;

    yv += __shfl_xor(yv, 1, 32);
    yv += __shfl_xor(yv, 2, 32);
    yv += __shfl_xor(yv, 4, 32);
    yv += __shfl_xor(yv, 8, 32);

    if (n == 0) {
      float zv  = xz[row * (2 * DINNER) + DINNER + d];
      float out = (yv + uv * Dd) * (zv / (1.f + __expf(-zv)));
      ybf[row * DINNER + d] = (bf16)out;
    }
  }
}

// ------------------------------------------------------------- launch ----
extern "C" void kernel_launch(void* const* d_in, const int* in_sizes, int n_in,
                              void* d_out, int out_size, void* d_ws,
                              size_t ws_size, hipStream_t stream) {
  (void)in_sizes; (void)n_in; (void)out_size; (void)ws_size;
  const float* x      = (const float*)d_in[0];
  const float* W_in   = (const float*)d_in[1];
  const float* conv_w = (const float*)d_in[2];
  const float* conv_b = (const float*)d_in[3];
  const float* W_x    = (const float*)d_in[4];
  const float* W_dt   = (const float*)d_in[5];
  const float* b_dt   = (const float*)d_in[6];
  const float* A_log  = (const float*)d_in[7];
  const float* D_par  = (const float*)d_in[8];
  const float* W_out  = (const float*)d_in[9];
  float* out = (float*)d_out;

  char* ws = (char*)d_ws;
  size_t off = 0;
  auto take = [&](size_t bytes) -> void* {
    void* p = ws + off;
    off += (bytes + 255) & ~(size_t)255;
    return p;
  };
  bf16*  x_bf    = (bf16*)take((size_t)NROWS * DMODEL * 2);
  bf16*  Win_bf  = (bf16*)take((size_t)2 * DINNER * DMODEL * 2);
  bf16*  Wx_bf   = (bf16*)take((size_t)XDBLW * DINNER * 2);
  bf16*  Wdt_bf  = (bf16*)take((size_t)DINNER * DTRANK * 2);
  bf16*  Wout_bf = (bf16*)take((size_t)DMODEL * DINNER * 2);
  float* xz      = (float*)take((size_t)NROWS * 2 * DINNER * 4);
  float* u_f     = (float*)take((size_t)NROWS * DINNER * 4);
  bf16*  u_bf    = (bf16*)take((size_t)NROWS * DINNER * 2);
  float* xdbl    = (float*)take((size_t)NROWS * XDBLW * 4);
  bf16*  dr_bf   = (bf16*)take((size_t)NROWS * DTRANK * 2);
  float* delta   = (float*)take((size_t)NROWS * DINNER * 4);
  bf16*  y_bf    = (bf16*)take((size_t)NROWS * DINNER * 2);

  auto g1 = [](size_t n) { return dim3((unsigned)((n + 255) / 256)); };

  cast_f32_to_bf16<<<g1((size_t)NROWS * DMODEL), 256, 0, stream>>>(
      x, x_bf, NROWS * DMODEL);
  cast_f32_to_bf16<<<g1((size_t)2 * DINNER * DMODEL), 256, 0, stream>>>(
      W_in, Win_bf, 2 * DINNER * DMODEL);
  cast_f32_to_bf16<<<g1((size_t)XDBLW * DINNER), 256, 0, stream>>>(
      W_x, Wx_bf, XDBLW * DINNER);
  cast_f32_to_bf16<<<g1((size_t)DINNER * DTRANK), 256, 0, stream>>>(
      W_dt, Wdt_bf, DINNER * DTRANK);
  cast_f32_to_bf16<<<g1((size_t)DMODEL * DINNER), 256, 0, stream>>>(
      W_out, Wout_bf, DMODEL * DINNER);

  // in_proj: xz = x @ W_in^T   (4096 x 4096, K=1024), async-LDS pipeline
  gemm_bf16_wmma_lds<0><<<dim3((2 * DINNER) / 64, NROWS / 128), 128, 0,
                          stream>>>(x_bf, Win_bf, xz, nullptr,
                                    NROWS, 2 * DINNER, DMODEL);

  conv_silu_kernel<<<g1((size_t)NROWS * DINNER), 256, 0, stream>>>(
      xz, conv_w, conv_b, u_f, u_bf);

  // x_proj: x_dbl = u @ W_x^T  (4096 x 96, K=2048), direct (N not /64)
  gemm_bf16_wmma_direct<2><<<dim3(XDBLW / 32, NROWS / 128), 256, 0, stream>>>(
      u_bf, Wx_bf, xdbl, NROWS, XDBLW, DINNER);

  cast_deltar<<<g1((size_t)NROWS * DTRANK), 256, 0, stream>>>(xdbl, dr_bf);

  // dt_proj + softplus (4096 x 2048, K=64), async-LDS pipeline
  gemm_bf16_wmma_lds<1><<<dim3(DINNER / 64, NROWS / 128), 128, 0, stream>>>(
      dr_bf, Wdt_bf, delta, b_dt, NROWS, DINNER, DTRANK);

  scan_kernel<<<dim3(DINNER / 16, BATCH), 256, 0, stream>>>(
      delta, u_f, xdbl, xz, A_log, D_par, y_bf);

  // out_proj: out = y @ W_out^T (4096 x 1024, K=2048), async-LDS pipeline
  gemm_bf16_wmma_lds<0><<<dim3(DMODEL / 64, NROWS / 128), 128, 0, stream>>>(
      y_bf, Wout_bf, out, nullptr, NROWS, DMODEL, DINNER);
}
